// mLSTMBlock_86543591014855
// MI455X (gfx1250) — matrix operand order repaired and checked by reading, not compiled, not run on hardware
//
#include <hip/hip_runtime.h>
#include <hip/hip_bf16.h>
#include <math.h>
#include <stdint.h>

typedef __bf16 bf16;
typedef __attribute__((ext_vector_type(16))) __bf16 v16bf;
typedef __attribute__((ext_vector_type(8)))  __bf16 bf16x8;
typedef __attribute__((ext_vector_type(8)))  float  v8f;
typedef int v4i __attribute__((vector_size(4 * sizeof(int))));

// Problem constants (from reference)
#define BB 2
#define LL 1536
#define DD 1024
#define HH 8
#define CC 2048            // 2*D
#define DPH 256            // C/H
#define RR (BB*LL)         // 3072 rows

#define AS(n) __attribute__((address_space(n)))

#if __has_builtin(__builtin_amdgcn_global_load_async_to_lds_b128)
#define HAVE_ASYNC_LDS 1
#else
#define HAVE_ASYNC_LDS 0
#endif

// One 16-byte global->LDS copy. Async (ASYNCcnt-tracked, no VGPR round-trip)
// when the toolchain exposes the gfx1250 builtin; plain load/store otherwise.
static __device__ __forceinline__ void async_b128(const bf16* g, bf16* l)
{
#if HAVE_ASYNC_LDS
    __builtin_amdgcn_global_load_async_to_lds_b128(
        (AS(1) v4i*)(uintptr_t)g, (AS(3) v4i*)(uintptr_t)l, 0, 0);
#else
    *(bf16x8*)l = *(const bf16x8*)g;
#endif
}

static __device__ __forceinline__ void wait_async_lds()
{
#if HAVE_ASYNC_LDS
#if __has_builtin(__builtin_amdgcn_s_wait_asynccnt)
    __builtin_amdgcn_s_wait_asynccnt(0);
#else
    asm volatile("s_wait_asynccnt 0x0" ::: "memory");
#endif
#endif
}

static __device__ __forceinline__ v16bf cat8(bf16x8 lo, bf16x8 hi)
{
    return __builtin_shufflevector(lo, hi, 0, 1, 2, 3, 4, 5, 6, 7,
                                   8, 9, 10, 11, 12, 13, 14, 15);
}

// ---------------------------------------------------------------------------
// Tiled bf16 WMMA GEMM:  C[z] = A[z] * op(B[z]) (+bias)(+resid)(+=C)
// Tile 64x128, K-step 32, LDS double-buffered: async-copy K-step k+1 into
// buffer buf^1 while the 4 WMMAs consume buffer buf; s_wait_asynccnt + barrier
// rotates buffers. 8 waves in 4x2 grid; each wave owns 16x64 via four 16x16
// f32 accumulators sharing one A fragment.
// REQUIRES: M % 64 == 0, N % 128 == 0, K % 32 == 0 (true for all calls here).
// ---------------------------------------------------------------------------
#define TBM 64
#define TBN 128
#define TBK 32

__global__ __launch_bounds__(256)
void gemm_bf16_wmma(const bf16* __restrict__ A, long long strideAb, int lda,
                    const bf16* __restrict__ Bp, long long strideBb, int ldb, int transB,
                    float* __restrict__ Cp, long long strideCb, int ldc,
                    const float* __restrict__ bias, int biasStride,
                    const float* __restrict__ resid, int ldr,
                    int M, int N, int Kd, int accumulate)
{
    __shared__ bf16 sA [2][TBM * TBK];    // row-major: row stride 32 (64 B)
    __shared__ bf16 sBt[2][TBN * TBK];    // n-major:  row stride 32 (64 B)

    const int z = blockIdx.z;
    A  += (long long)z * strideAb;
    Bp += (long long)z * strideBb;
    Cp += (long long)z * strideCb;
    const float* biasZ = bias ? (bias + (long long)z * biasStride) : nullptr;

    const int tid  = threadIdx.x;
    const int lane = tid & 31;
    const int wave = tid >> 5;      // 0..7
    const int wr   = wave & 3;      // M sub-tile (16 rows)
    const int wc   = wave >> 2;     // N half (64 cols)
    const int m0   = blockIdx.x * TBM;
    const int n0   = blockIdx.y * TBN;

    // staging coordinates (fixed per thread)
    const int ar  = tid >> 2;               // A row, 0..63
    const int ac8 = (tid & 3) << 3;         // A col chunk

    v8f acc[4] = {};

    auto stage = [&](int buf, int k0) {
        // A tile (64x32): one b128 async copy per thread
        async_b128(A + (size_t)(m0 + ar) * lda + k0 + ac8,
                   &sA[buf][ar * TBK + ac8]);
        if (transB) {
            // B is (N x K) row-major: contiguous along k -> pure async copies
#pragma unroll
            for (int t = 0; t < 2; ++t) {
                const int idx = tid + t * 256;   // 0..511
                const int n  = idx >> 2;
                const int k8 = (idx & 3) << 3;
                async_b128(Bp + (size_t)(n0 + n) * ldb + k0 + k8,
                           &sBt[buf][n * TBK + k8]);
            }
        } else {
            // B is (K x N) row-major: b128 in, scatter-transpose out
#pragma unroll
            for (int t = 0; t < 2; ++t) {
                const int idx = tid + t * 256;   // 0..511
                const int k  = idx >> 4;
                const int n8 = (idx & 15) << 3;
                bf16x8 v = *(const bf16x8*)(Bp + (size_t)(k0 + k) * ldb + n0 + n8);
#pragma unroll
                for (int j = 0; j < 8; ++j)
                    sBt[buf][(n8 + j) * TBK + k] = v[j];
            }
        }
    };

    stage(0, 0);
    wait_async_lds();
    __syncthreads();

    for (int k0 = 0; k0 < Kd; k0 += TBK) {
        const int buf = (k0 / TBK) & 1;
        if (k0 + TBK < Kd) {
            stage(buf ^ 1, k0 + TBK);
            __builtin_prefetch(A + (size_t)(m0 + ar) * lda + k0 + 2 * TBK + ac8, 0, 1);
        }

        // ---- fragments + 4 WMMAs from buffer `buf` -----------------------
        {
            const bf16* aRow = &sA[buf][(wr * 16 + (lane & 15)) * TBK];
            const int   kg   = (lane >> 4) << 4;
            v16bf afrag = cat8(*(const bf16x8*)(aRow + kg),
                               *(const bf16x8*)(aRow + kg + 8));
            const int kb = (lane >> 4) << 4;
#pragma unroll
            for (int s = 0; s < 4; ++s) {
                const bf16* bRow = &sBt[buf][(wc * 64 + s * 16 + (lane & 15)) * TBK] + kb;
                v16bf bfrag = cat8(*(const bf16x8*)(bRow),
                                   *(const bf16x8*)(bRow + 8));
                acc[s] = __builtin_amdgcn_wmma_f32_16x16x32_bf16(
                    false, afrag, false, bfrag, (short)0, acc[s], false, false);
            }
        }
        wait_async_lds();
        __syncthreads();
    }

    // ---- store: M = r + 8*(lane>>4), N = lane&15 per 16x16 sub-tile -------
    const int moff = wr * 16 + ((lane >> 4) << 3);
#pragma unroll
    for (int s = 0; s < 4; ++s) {
        const int gn = n0 + wc * 64 + s * 16 + (lane & 15);
#pragma unroll
        for (int r = 0; r < 8; ++r) {
            const int gm = m0 + moff + r;
            float v = acc[s][r];
            if (biasZ) v += biasZ[gn];
            if (resid) v += resid[(size_t)gm * ldr + gn];
            const size_t idx = (size_t)gm * ldc + gn;
            if (accumulate) v += Cp[idx];
            Cp[idx] = v;
        }
    }
}

// ---------------------------------------------------------------------------
// Elementwise / small kernels
// ---------------------------------------------------------------------------
__global__ __launch_bounds__(256)
void convert2d_f32_bf16(const float* __restrict__ src, int sld,
                        bf16* __restrict__ dst, int dld,
                        int rows, int cols, float mul)
{
    size_t n = (size_t)rows * cols;
    for (size_t i = (size_t)blockIdx.x * 256 + threadIdx.x; i < n;
         i += (size_t)gridDim.x * 256) {
        int r = (int)(i / cols), c = (int)(i % cols);
        dst[(size_t)r * dld + c] = (bf16)(src[(size_t)r * sld + c] * mul);
    }
}

// Wk[ci*C + co] = conv_w[(co*C + ci)*K + kidx]  (bf16)
__global__ __launch_bounds__(256)
void extract_convw(const float* __restrict__ cw, int kidx, bf16* __restrict__ dst)
{
    size_t n = (size_t)CC * CC;
    for (size_t i = (size_t)blockIdx.x * 256 + threadIdx.x; i < n;
         i += (size_t)gridDim.x * 256) {
        int ci = (int)(i / CC), co = (int)(i % CC);
        dst[i] = (bf16)cw[((size_t)co * CC + ci) * 4 + kidx];
    }
}

__global__ __launch_bounds__(256)
void ln_rows(const float* __restrict__ x, const float* __restrict__ w,
             const float* __restrict__ b, bf16* __restrict__ out, int cols)
{
    const int row = blockIdx.x, tid = threadIdx.x;
    const float* xr = x + (size_t)row * cols;
    __shared__ float red[256];
    float s = 0.f;
    for (int c = tid; c < cols; c += 256) s += xr[c];
    red[tid] = s; __syncthreads();
    for (int o = 128; o > 0; o >>= 1) { if (tid < o) red[tid] += red[tid + o]; __syncthreads(); }
    const float mu = red[0] / cols;
    __syncthreads();
    float v = 0.f;
    for (int c = tid; c < cols; c += 256) { float d = xr[c] - mu; v += d * d; }
    red[tid] = v; __syncthreads();
    for (int o = 128; o > 0; o >>= 1) { if (tid < o) red[tid] += red[tid + o]; __syncthreads(); }
    const float rstd = rsqrtf(red[0] / cols + 1e-5f);
    for (int c = tid; c < cols; c += 256)
        out[(size_t)row * cols + c] = (bf16)((xr[c] - mu) * rstd * w[c] + (b ? b[c] : 0.f));
}

__global__ __launch_bounds__(256)
void silu_store(float* __restrict__ qk, bf16* __restrict__ qkbf, size_t n)
{
    for (size_t i = (size_t)blockIdx.x * 256 + threadIdx.x; i < n;
         i += (size_t)gridDim.x * 256) {
        float v = qk[i];
        float s = v / (1.f + __expf(-v));
        qk[i] = s;
        qkbf[i] = (bf16)s;
    }
}

// proj rows [h*512 + {q:0..255,k:256..511}], v rows [h*256 + d]
// -> Q/K/V as [b][h][l][256] bf16 ; K pre-scaled by 1/sqrt(DPH)
__global__ __launch_bounds__(256)
void make_qkv(const float* __restrict__ proj, const float* __restrict__ vbuf,
              bf16* __restrict__ Q, bf16* __restrict__ Kb, bf16* __restrict__ V)
{
    size_t n = (size_t)BB * HH * LL * DPH;
    for (size_t i = (size_t)blockIdx.x * 256 + threadIdx.x; i < n;
         i += (size_t)gridDim.x * 256) {
        int d = (int)(i & (DPH - 1));
        int l = (int)((i / DPH) % LL);
        int h = (int)((i / ((size_t)DPH * LL)) % HH);
        int b = (int)(i / ((size_t)DPH * LL * HH));
        size_t row = (size_t)b * LL + l;
        Q[i]  = (bf16)proj[row * 4096 + h * 512 + d];
        Kb[i] = (bf16)(proj[row * 4096 + h * 512 + 256 + d] * 0.0625f); // 1/16
        V[i]  = (bf16)vbuf[row * 2048 + h * 256 + d];
    }
}

// gates[row][g] = dot(qkv_row, wif[g]) + bif[g]; g<8 -> exp, g>=8 -> logsigmoid
__global__ __launch_bounds__(128)
void gates_kernel(const float* __restrict__ proj, const float* __restrict__ vbuf,
                  const float* __restrict__ wif, const float* __restrict__ bif,
                  float* __restrict__ ig, float* __restrict__ fls)
{
    const int row = blockIdx.x, g = blockIdx.y, tid = threadIdx.x;
    const float* wrow = wif + (size_t)g * (3 * CC);
    float s = 0.f;
    for (int j = tid; j < 3 * CC; j += 128) {
        float q = (j < 4096) ? proj[(size_t)row * 4096 + j]
                             : vbuf[(size_t)row * 2048 + (j - 4096)];
        s += q * wrow[j];
    }
    __shared__ float red[128];
    red[tid] = s; __syncthreads();
    for (int o = 64; o > 0; o >>= 1) { if (tid < o) red[tid] += red[tid + o]; __syncthreads(); }
    if (tid == 0) {
        float val = red[0] + bif[g];
        int b = row / LL, l = row % LL;
        if (g < HH) {
            ig[((size_t)(b * HH + g)) * LL + l] = __expf(val);
        } else {
            float x = val;
            fls[((size_t)(b * HH + (g - HH))) * LL + l] =
                fminf(x, 0.f) - log1pf(__expf(-fabsf(x)));
        }
    }
}

__global__ __launch_bounds__(32)
void cumsum_f(const float* __restrict__ fls, float* __restrict__ fc)
{
    int t = threadIdx.x;
    if (t < BB * HH) {
        float run = 0.f;
        for (int l = 0; l < LL; ++l) {
            run += fls[(size_t)t * LL + l];
            fc[(size_t)t * LL + l] = run;
        }
    }
}

// ---------------------------------------------------------------------------
// Flash-style mLSTM core.  Block = one (b, h, 16-row output tile).
// Streams 32-wide i-chunks; waves 0/1 build decayed P via WMMA S=Q.K^T,
// all 8 waves accumulate P.V over their 32 output columns with WMMA.
// Q/K tiles staged with async global->LDS copies; V staged transposed
// (sVt[d][i]) so P.V B-fragments are ds_load_b128.
// ---------------------------------------------------------------------------
__global__ __launch_bounds__(256)
void mlstm_core(const bf16* __restrict__ Q, const bf16* __restrict__ Kb,
                const bf16* __restrict__ V, const float* __restrict__ ig,
                const float* __restrict__ fc, float* __restrict__ hOut)
{
    __shared__ bf16  sQ [16][DPH];     // 512 B rows
    __shared__ bf16  sK [32][DPH];     // 512 B rows
    __shared__ bf16  sVt[DPH][32];     // transposed, 64 B rows
    __shared__ bf16  sP [16][32];      // 64 B rows
    __shared__ float sFCo[16], sFCi[32], sIGi[32];
    __shared__ float sMnew[16], sScale[16], sM[16], sBsum[16], sRS[2][16];

    const int tid  = threadIdx.x;
    const int lane = tid & 31;
    const int wave = tid >> 5;
    const int o0   = blockIdx.x * 16;
    const int h    = blockIdx.y;
    const int b    = blockIdx.z;
    const size_t qbase  = ((size_t)(b * HH + h) * LL) * DPH;
    const size_t fcbase = (size_t)(b * HH + h) * LL;

    // Load Q tile (async b128 chunks), init running stats
    for (int i = tid; i < 16 * DPH / 8; i += 256) {
        int r  = i >> 5;            // 32 chunks per 256-wide row
        int c8 = (i & 31) << 3;
        async_b128(Q + qbase + (size_t)(o0 + r) * DPH + c8, &sQ[r][c8]);
    }
    if (tid < 16) {
        sM[tid] = -__builtin_inff();
        sBsum[tid] = 0.f;
        sFCo[tid] = fc[fcbase + o0 + tid];
    }
    wait_async_lds();
    __syncthreads();

    v8f acc0 = {};
    v8f acc1 = {};

    for (int i0 = 0; i0 < o0 + 16; i0 += 32) {
        // (a) stage K (async, row-major) + V (transposed) chunk + gate scalars
        for (int i = tid; i < 32 * DPH / 8; i += 256) {
            int r  = i >> 5;
            int c8 = (i & 31) << 3;
            async_b128(Kb + qbase + (size_t)(i0 + r) * DPH + c8, &sK[r][c8]);
            bf16x8 vv = *(const bf16x8*)(V + qbase + (size_t)(i0 + r) * DPH + c8);
#pragma unroll
            for (int j = 0; j < 8; ++j) sVt[c8 + j][r] = vv[j];
        }
        if (tid < 32) {
            sFCi[tid] = fc[fcbase + i0 + tid];
            sIGi[tid] = ig[fcbase + i0 + tid];
        }
        wait_async_lds();
        __syncthreads();

        // (b) running max / rescale factor per output row (wave 0)
        if (wave == 0 && lane < 16) {
            const int go = o0 + lane;
            const float fco = sFCo[lane];
            float mc = -__builtin_inff();
            for (int i = 0; i < 32; ++i)
                if (i0 + i <= go) mc = fmaxf(mc, fco - sFCi[i] + sIGi[i]);
            float mold = sM[lane];
            float mnew = fmaxf(mold, mc);
            sMnew[lane]  = mnew;
            sScale[lane] = (mold == -__builtin_inff()) ? 0.f : __expf(mold - mnew);
            sM[lane] = mnew;
        }
        __syncthreads();

        // (c) waves 0/1: S = Q.K^T via 8 WMMAs, decay+mask -> P (bf16 in LDS)
        if (wave < 2) {
            v8f s = {};
            const int kg  = (lane >> 4) << 4;
            const int kb2 = (lane >> 4) << 4;
            const bf16* qRow = &sQ[lane & 15][0];
            const bf16* kRow = &sK[wave * 16 + (lane & 15)][0];
#pragma unroll
            for (int d0 = 0; d0 < DPH; d0 += 32) {
                v16bf aq = cat8(*(const bf16x8*)(qRow + d0 + kg),
                                *(const bf16x8*)(qRow + d0 + kg + 8));
                v16bf bk = cat8(*(const bf16x8*)(kRow + d0 + kb2),
                                *(const bf16x8*)(kRow + d0 + kb2 + 8));
                s = __builtin_amdgcn_wmma_f32_16x16x32_bf16(false, aq, false, bk,
                                                            (short)0, s, false, false);
            }
            float psum[8];
#pragma unroll
            for (int r = 0; r < 8; ++r) {
                const int Mr = r + ((lane >> 4) << 3);
                const int il = wave * 16 + (lane & 15);
                const int gi = i0 + il, go = o0 + Mr;
                float p = 0.f;
                if (gi <= go)
                    p = s[r] * __expf(sFCo[Mr] - sFCi[il] + sIGi[il] - sMnew[Mr]);
                sP[Mr][il] = (bf16)p;
                psum[r] = p;
            }
#pragma unroll
            for (int off = 1; off < 16; off <<= 1)
#pragma unroll
                for (int r = 0; r < 8; ++r)
                    psum[r] += __shfl_xor(psum[r], off, 32);
            if ((lane & 15) == 0)
#pragma unroll
                for (int r = 0; r < 8; ++r)
                    sRS[wave][r + ((lane >> 4) << 3)] = psum[r];
        }
        __syncthreads();

        // (d) all waves: rescale accumulators, then acc += P.V (2 WMMAs)
#pragma unroll
        for (int r = 0; r < 8; ++r) {
            float sc = sScale[r + ((lane >> 4) << 3)];
            acc0[r] *= sc;
            acc1[r] *= sc;
        }
        {
            const int kg  = (lane >> 4) << 4;
            const int kb2 = (lane >> 4) << 4;
            const bf16* pRow = &sP[lane & 15][0];
            v16bf ap = cat8(*(const bf16x8*)(pRow + kg),
                            *(const bf16x8*)(pRow + kg + 8));
            const bf16* vRow0 = &sVt[wave * 32 + (lane & 15)][0] + kb2;
            const bf16* vRow1 = &sVt[wave * 32 + 16 + (lane & 15)][0] + kb2;
            v16bf bv0 = cat8(*(const bf16x8*)(vRow0), *(const bf16x8*)(vRow0 + 8));
            v16bf bv1 = cat8(*(const bf16x8*)(vRow1), *(const bf16x8*)(vRow1 + 8));
            acc0 = __builtin_amdgcn_wmma_f32_16x16x32_bf16(false, ap, false, bv0,
                                                           (short)0, acc0, false, false);
            acc1 = __builtin_amdgcn_wmma_f32_16x16x32_bf16(false, ap, false, bv1,
                                                           (short)0, acc1, false, false);
        }
        if (wave == 0 && lane < 16)
            sBsum[lane] = sBsum[lane] * sScale[lane] + sRS[0][lane] + sRS[1][lane];
        __syncthreads();
    }

    // Finalize: n = max(|bsum|, exp(-m)); h = acc / n
#pragma unroll
    for (int r = 0; r < 8; ++r) {
        const int Mr = r + ((lane >> 4) << 3);
        const float nrm = fmaxf(fabsf(sBsum[Mr]), __expf(-sM[Mr]));
        const float inv = 1.f / nrm;
        const size_t base = qbase + (size_t)(o0 + Mr) * DPH;
        const int dc = wave * 32 + (lane & 15);
        hOut[base + dc]      = acc0[r] * inv;
        hOut[base + dc + 16] = acc1[r] * inv;
    }
}

// Fused head-LN + skip*qk + SiLU(x2) gate -> xm (bf16)
__global__ __launch_bounds__(256)
void headnorm_fuse(const float* __restrict__ hOut, const float* __restrict__ hnw,
                   const float* __restrict__ skip, const float* __restrict__ qk,
                   const float* __restrict__ y, bf16* __restrict__ xmbf)
{
    const int row  = blockIdx.x;
    const int b    = row / LL, l = row % LL;
    const int wave = threadIdx.x >> 5, lane = threadIdx.x & 31;
    const float* hr = hOut + ((size_t)(b * HH + wave) * LL + l) * DPH;

    float vals[8];
    float s = 0.f;
#pragma unroll
    for (int j = 0; j < 8; ++j) { vals[j] = hr[lane + 32 * j]; s += vals[j]; }
#pragma unroll
    for (int off = 1; off < 32; off <<= 1) s += __shfl_xor(s, off, 32);
    const float mu = s * (1.f / DPH);
    float v = 0.f;
#pragma unroll
    for (int j = 0; j < 8; ++j) { float d = vals[j] - mu; v += d * d; }
#pragma unroll
    for (int off = 1; off < 32; off <<= 1) v += __shfl_xor(v, off, 32);
    const float rstd = rsqrtf(v * (1.f / DPH) + 1e-5f);

#pragma unroll
    for (int j = 0; j < 8; ++j) {
        const int d = lane + 32 * j;
        const int c = wave * DPH + d;
        float hn  = (vals[j] - mu) * rstd * hnw[d];
        float x1o = hn + skip[c] * qk[(size_t)row * CC + c];
        float x2v = y[(size_t)row * 4096 + CC + c];
        float sil = x2v / (1.f + __expf(-x2v));
        xmbf[(size_t)row * CC + c] = (bf16)(x1o * sil);
    }
}

// ---------------------------------------------------------------------------
// Host-side orchestration
// ---------------------------------------------------------------------------
extern "C" void kernel_launch(void* const* d_in, const int* in_sizes, int n_in,
                              void* d_out, int out_size, void* d_ws, size_t ws_size,
                              hipStream_t stream)
{
    const float* x      = (const float*)d_in[0];
    const float* ln_w   = (const float*)d_in[1];
    const float* ln_b   = (const float*)d_in[2];
    const float* w1     = (const float*)d_in[3];
    const float* b1     = (const float*)d_in[4];
    const float* conv_w = (const float*)d_in[5];
    const float* conv_b = (const float*)d_in[6];
    const float* skip   = (const float*)d_in[7];
    const float* wqk    = (const float*)d_in[8];
    const float* bqk    = (const float*)d_in[9];
    const float* wv     = (const float*)d_in[10];
    const float* bv     = (const float*)d_in[11];
    const float* wif    = (const float*)d_in[12];
    const float* bif    = (const float*)d_in[13];
    const float* hn_w   = (const float*)d_in[14];
    const float* w2     = (const float*)d_in[15];
    const float* b2     = (const float*)d_in[16];
    float* out = (float*)d_out;

    char* wsp = (char*)d_ws;
    size_t off = 0;
    auto alloc = [&](size_t bytes) -> void* {
        void* p = wsp + off;
        off += (bytes + 255) & ~(size_t)255;
        return p;
    };

    bf16*  lnbf   = (bf16*) alloc((size_t)RR * DD * 2);
    bf16*  w1bf   = (bf16*) alloc((size_t)4 * DD * DD * 2);
    bf16*  w2bf   = (bf16*) alloc((size_t)DD * CC * 2);
    bf16*  wqkbf  = (bf16*) alloc((size_t)HH * DPH * 2 * DPH * 2);
    bf16*  wvbf   = (bf16*) alloc((size_t)HH * DPH * DPH * 2);
    bf16*  cw2    = (bf16*) alloc((size_t)CC * CC * 2);
    bf16*  cw3    = (bf16*) alloc((size_t)CC * CC * 2);
    float* y      = (float*)alloc((size_t)RR * 4096 * 4);
    bf16*  x1bf   = (bf16*) alloc((size_t)RR * CC * 2);
    float* qk     = (float*)alloc((size_t)RR * CC * 4);
    bf16*  qkbf   = (bf16*) alloc((size_t)RR * CC * 2);
    float* proj   = (float*)alloc((size_t)RR * 4096 * 4);
    float* vbuf   = (float*)alloc((size_t)RR * CC * 4);
    bf16*  Qbf    = (bf16*) alloc((size_t)RR * CC * 2);
    bf16*  Kbf    = (bf16*) alloc((size_t)RR * CC * 2);
    bf16*  Vbf    = (bf16*) alloc((size_t)RR * CC * 2);
    float* ig     = (float*)alloc((size_t)BB * HH * LL * 4);
    float* fls    = (float*)alloc((size_t)BB * HH * LL * 4);
    float* fc     = (float*)alloc((size_t)BB * HH * LL * 4);
    float* hOut   = (float*)alloc((size_t)RR * CC * 4);
    bf16*  xmbf   = (bf16*) alloc((size_t)RR * CC * 2);
    (void)ws_size; (void)n_in; (void)in_sizes; (void)out_size;

    // 1) weight conversions
    convert2d_f32_bf16<<<1024, 256, 0, stream>>>(w1, DD, w1bf, DD, 4 * DD, DD, 1.f);
    convert2d_f32_bf16<<<1024, 256, 0, stream>>>(w2, CC, w2bf, CC, DD, CC, 1.f);
    convert2d_f32_bf16<<<512, 256, 0, stream>>>(wqk, 2 * DPH, wqkbf, 2 * DPH, HH * DPH, 2 * DPH, 1.f);
    convert2d_f32_bf16<<<512, 256, 0, stream>>>(wv, DPH, wvbf, DPH, HH * DPH, DPH, 1.f);
    extract_convw<<<2048, 256, 0, stream>>>(conv_w, 2, cw2);
    extract_convw<<<2048, 256, 0, stream>>>(conv_w, 3, cw3);

    // 2) LN(x) -> bf16
    ln_rows<<<RR, 256, 0, stream>>>(x, ln_w, ln_b, lnbf, DD);

    // 3) y = LN(x) @ w1^T + b1   (3072 x 4096 x 1024)
    gemm_bf16_wmma<<<dim3(RR / TBM, 4096 / TBN, 1), 256, 0, stream>>>(
        lnbf, 0, DD, w1bf, 0, DD, 1, y, 0, 4096,
        b1, 0, nullptr, 0, RR, 4096, DD, 0);

    // 4) x1 -> bf16 (columns [0,2048) of y)
    convert2d_f32_bf16<<<2048, 256, 0, stream>>>(y, 4096, x1bf, CC, RR, CC, 1.f);

    // 5) batch-axis causal conv == tap GEMMs
    gemm_bf16_wmma<<<dim3(LL / TBM, CC / TBN, 1), 256, 0, stream>>>(
        x1bf, 0, CC, cw3, 0, CC, 0, qk, 0, CC,
        conv_b, 0, nullptr, 0, LL, CC, CC, 0);
    gemm_bf16_wmma<<<dim3(LL / TBM, CC / TBN, 1), 256, 0, stream>>>(
        x1bf, 0, CC, cw2, 0, CC, 0, qk + (size_t)LL * CC, 0, CC,
        conv_b, 0, nullptr, 0, LL, CC, CC, 0);
    gemm_bf16_wmma<<<dim3(LL / TBM, CC / TBN, 1), 256, 0, stream>>>(
        x1bf + (size_t)LL * CC, 0, CC, cw3, 0, CC, 0, qk + (size_t)LL * CC, 0, CC,
        nullptr, 0, nullptr, 0, LL, CC, CC, 1);

    // 6) qk = silu(qk), keep f32 + bf16
    silu_store<<<2048, 256, 0, stream>>>(qk, qkbf, (size_t)RR * CC);

    // 7) per-head projections (batched over heads via grid.z)
    gemm_bf16_wmma<<<dim3(RR / TBM, 512 / TBN, HH), 256, 0, stream>>>(
        qkbf, DPH, CC, wqkbf, (long long)DPH * 2 * DPH, 2 * DPH, 0,
        proj, 2 * DPH, 4096, bqk, 2 * DPH, nullptr, 0, RR, 2 * DPH, DPH, 0);
    gemm_bf16_wmma<<<dim3(RR / TBM, 256 / TBN, HH), 256, 0, stream>>>(
        x1bf, DPH, CC, wvbf, (long long)DPH * DPH, DPH, 0,
        vbuf, DPH, CC, bv, DPH, nullptr, 0, RR, DPH, DPH, 0);

    // 8) gates, cumsum, QKV rearrange
    gates_kernel<<<dim3(RR, 2 * HH, 1), 128, 0, stream>>>(proj, vbuf, wif, bif, ig, fls);
    cumsum_f<<<1, 32, 0, stream>>>(fls, fc);
    make_qkv<<<2048, 256, 0, stream>>>(proj, vbuf, Qbf, Kbf, Vbf);

    // 9) mLSTM attention core (flash-style)
    mlstm_core<<<dim3(LL / 16, HH, BB), 256, 0, stream>>>(Qbf, Kbf, Vbf, ig, fc, hOut);

    // 10) head-LN + skip + SiLU gate -> xm (bf16)
    headnorm_fuse<<<RR, 256, 0, stream>>>(hOut, hn_w, skip, qk, y, xmbf);

    // 11) out = xm @ w2^T + b2 + x
    gemm_bf16_wmma<<<dim3(RR / TBM, DD / TBN, 1), 256, 0, stream>>>(
        xmbf, 0, CC, w2bf, 0, CC, 1, out, 0, DD,
        b2, 0, x, DD, RR, DD, CC, 0);
}
